// DynamicalRNN_34291018891726
// MI455X (gfx1250) — matrix operand order repaired and checked by reading, not compiled, not run on hardware
//
#include <hip/hip_runtime.h>
#include <hip/hip_bf16.h>

// MI455X (gfx1250, wave32) RNN forward:
//   h_t = tanh(x_t @ Wi^T + h_{t-1} @ Wh^T),  y_t = h_t @ Wy^T
// B=128, T=1024, NX=256, NH=512, NY=256.
//
//   Phase 1: A = x @ Wi^T  (parallel bf16 WMMA GEMM, f32 accum, async-LDS staging)
//   Phase 2: serial scan; 8 persistent WGs (one per 16-batch-row slab),
//            h ping-ponged in LDS as WMMA A-fragments; Abuf double-buffered in LDS
//            via GLOBAL_LOAD_ASYNC_TO_LDS_B128; Wh streamed from L2.
//   Phase 3: y = hs @ Wy^T (parallel bf16 WMMA GEMM, async-LDS staging)

#define B_   128
#define T_   1024
#define NX_  256
#define NH_  512
#define NY_  256

typedef __attribute__((ext_vector_type(16))) __bf16 v16bf;
typedef __attribute__((ext_vector_type(8)))  float  v8f;

// ---- WMMA bf16 16x16x32 fragment layouts (CDNA5 ISA, 7.12.2) ----
__device__ __forceinline__ void afrag_from(int lane, int e, int& m, int& kk) {
    m  = lane & 15;
    kk = (e & 7) + ((e & 8) ? 16 : 0) + ((lane & 16) ? 8 : 0);
}
__device__ __forceinline__ void afrag_to(int m, int kk, int& lane, int& e) {
    lane = m + ((kk & 8) ? 16 : 0);
    e    = (kk & 7) | ((kk & 16) ? 8 : 0);
}
// B (32x16, KxN): n = lane&15, kk = (e&7) + (e&8 ? 16 : 0) + (lane>=16 ? 8 : 0)
// C/D (16x16 f32): acc[r] at lane -> m = r + (lane&16 ? 8:0), n = lane&15

#define WMMA_BF16(a, b, c) \
    __builtin_amdgcn_wmma_f32_16x16x32_bf16(false, (a), false, (b), (short)0, (c), false, false)

// Low 32 bits of a flat LDS pointer == LDS byte address (ISA 10.2).
__device__ __forceinline__ uint32_t lds_addr32(const void* p) {
    return (uint32_t)(uintptr_t)p;
}
// Async copy 16B global -> LDS, tracked by ASYNCcnt (CDNA5 ISA).
__device__ __forceinline__ void async_g2l_b128(uint32_t lds, uint64_t gaddr) {
    asm volatile("global_load_async_to_lds_b128 %0, %1, off"
                 :: "v"(lds), "v"(gaddr) : "memory");
}
__device__ __forceinline__ void wait_async0() {
    asm volatile("s_wait_asynccnt 0x0" ::: "memory");
}

// -------------------- packing kernels (fp32 -> bf16 fragments) --------------------

// xfrag layout: [rt = s*T + t][kt 0..7][lane 0..31][e 0..15]; s = batch slab (b/16)
__global__ __launch_bounds__(256) void pack_x_kernel(const float* __restrict__ x,
                                                     __bf16* __restrict__ xfrag) {
    size_t idx = (size_t)blockIdx.x * 256 + threadIdx.x;   // 2^25 elements
    int e    = (int)(idx & 15);
    int lane = (int)((idx >> 4) & 31);
    int kt   = (int)((idx >> 9) & 7);
    int t    = (int)((idx >> 12) & 1023);
    int s    = (int)(idx >> 22);
    int m, kk;
    afrag_from(lane, e, m, kk);
    int k = kt * 32 + kk;
    int b = s * 16 + m;
    xfrag[idx] = (__bf16)x[((size_t)b * T_ + t) * NX_ + k];
}

// W (N,K) row-major -> B-fragments: [nt][kt][lane][e], K = KT*32
__global__ __launch_bounds__(256) void pack_w_kernel(const float* __restrict__ W,
                                                     __bf16* __restrict__ Wp, int KT) {
    size_t idx = (size_t)blockIdx.x * 256 + threadIdx.x;
    int e    = (int)(idx & 15);
    int lane = (int)((idx >> 4) & 31);
    int frag = (int)(idx >> 9);
    int kt   = frag % KT;
    int nt   = frag / KT;
    int n    = nt * 16 + (lane & 15);
    int kk   = kt * 32 + (e & 7) + ((e & 8) ? 16 : 0) + ((lane & 16) ? 8 : 0);
    int K    = KT * 32;
    Wp[idx] = (__bf16)W[(size_t)n * K + kk];
}

// -------------------- phase 1: A = x @ Wi^T --------------------
__global__ __launch_bounds__(256) void gemm_xwi_kernel(const __bf16* __restrict__ xfrag,
                                                       const __bf16* __restrict__ Wip,
                                                       float* __restrict__ Abuf) {
    int rt = blockIdx.x;
    __shared__ __attribute__((aligned(32))) __bf16 aS[8 * 512];   // 8 KB: K=256 A row-tile
    {
        uint32_t l = lds_addr32(aS) + threadIdx.x * 16;
        uint64_t g = (uint64_t)(uintptr_t)(xfrag + (size_t)rt * 4096) + threadIdx.x * 16;
        async_g2l_b128(l, g);
        async_g2l_b128(l + 4096u, g + 4096u);
        wait_async0();
    }
    __syncthreads();

    int wave = threadIdx.x >> 5, lane = threadIdx.x & 31;
    v16bf a[8];
#pragma unroll
    for (int kt = 0; kt < 8; ++kt) a[kt] = *(const v16bf*)(aS + kt * 512 + lane * 16);

#pragma unroll
    for (int j = 0; j < 4; ++j) {                                  // 4 independent chains
        int nt = wave * 4 + j;
        v8f acc = {0.f, 0.f, 0.f, 0.f, 0.f, 0.f, 0.f, 0.f};
        const __bf16* bp = Wip + ((size_t)nt * 8) * 512 + (size_t)lane * 16;
#pragma unroll
        for (int kt = 0; kt < 8; ++kt) {
            v16bf b = *(const v16bf*)(bp + (size_t)kt * 512);
            acc = WMMA_BF16(a[kt], b, acc);
        }
        float* o = Abuf + ((size_t)rt * 32 + nt) * 256 + lane;     // tile layout, coalesced
#pragma unroll
        for (int r = 0; r < 8; ++r) o[r * 32] = acc[r];
    }
}

// -------------------- phase 2: serial recurrence --------------------
// grid = 8 (one WG per batch slab), 256 threads = 8 waves.
// LDS: h ping-pong (2x16KB) + Abuf double buffer (2x32KB) filled by async DMA.
__global__ __launch_bounds__(256) void rnn_recurrent_kernel(const __bf16* __restrict__ Whp,
                                                            const float* __restrict__ Abuf,
                                                            const float* __restrict__ h0,
                                                            float* __restrict__ hs_out,
                                                            __bf16* __restrict__ hfrag) {
    int s = blockIdx.x;
    __shared__ __attribute__((aligned(32))) __bf16 hbuf[2][16 * 512];   // 2 x 16 KB
    __shared__ __attribute__((aligned(16))) float  preS[2][32 * 256];   // 2 x 32 KB

    int wave = threadIdx.x >> 5, lane = threadIdx.x & 31;

    // stage Abuf tiles for t=0 (32 KB) with async DMA
    {
        uint64_t g = (uint64_t)(uintptr_t)(Abuf + (size_t)s * T_ * 32 * 256)
                   + threadIdx.x * 16;
        uint32_t l = lds_addr32(&preS[0][0]) + threadIdx.x * 16;
#pragma unroll
        for (int i = 0; i < 8; ++i)
            async_g2l_b128(l + (uint32_t)i * 4096u, g + (uint64_t)i * 4096u);
    }
    // init h_0 fragments from f32 input
    for (int i = threadIdx.x; i < 16 * 512; i += 256) {
        int e = i & 15, ln = (i >> 4) & 31, kt = i >> 9;
        int m, kk;
        afrag_from(ln, e, m, kk);
        hbuf[0][i] = (__bf16)h0[(size_t)(s * 16 + m) * NH_ + kt * 32 + kk];
    }
    wait_async0();
    __syncthreads();

    for (int t = 0; t < T_; ++t) {
        // kick off async staging of next step's pre-activations (overlaps WMMA work)
        if (t + 1 < T_) {
            uint64_t g = (uint64_t)(uintptr_t)(Abuf + ((size_t)s * T_ + (t + 1)) * 32 * 256)
                       + threadIdx.x * 16;
            uint32_t l = lds_addr32(&preS[(t + 1) & 1][0]) + threadIdx.x * 16;
#pragma unroll
            for (int i = 0; i < 8; ++i)
                async_g2l_b128(l + (uint32_t)i * 4096u, g + (uint64_t)i * 4096u);
        }

        const __bf16* hc = hbuf[t & 1];
        __bf16* hn = hbuf[(t + 1) & 1];
        const float* pr = preS[t & 1];
        size_t rt = (size_t)s * T_ + t;

        v16bf a[16];
#pragma unroll
        for (int kt = 0; kt < 16; ++kt)
            a[kt] = *(const v16bf*)(hc + kt * 512 + lane * 16);     // ds_load_b128 x2

#pragma unroll
        for (int j = 0; j < 4; ++j) {                               // 4 independent chains
            int nt = wave * 4 + j;
            v8f acc = {0.f, 0.f, 0.f, 0.f, 0.f, 0.f, 0.f, 0.f};
            const __bf16* bp = Whp + (size_t)nt * 16 * 512 + (size_t)lane * 16;
#pragma unroll
            for (int kt = 0; kt < 16; ++kt) {
                v16bf b = *(const v16bf*)(bp + (size_t)kt * 512);   // L2-resident Wh
                acc = WMMA_BF16(a[kt], b, acc);
            }
#pragma unroll
            for (int r = 0; r < 8; ++r) {
                float v = tanhf(acc[r] + pr[nt * 256 + r * 32 + lane]);
                int m = r + ((lane & 16) ? 8 : 0);
                int n = nt * 16 + (lane & 15);
                hs_out[((size_t)(s * 16 + m) * T_ + t) * NH_ + n] = v;
                __bf16 hv = (__bf16)v;
                int kt2 = n >> 5, kkl = n & 31;
                int l2, e2;
                afrag_to(m, kkl, l2, e2);
                hn[kt2 * 512 + l2 * 16 + e2] = hv;                  // next-step A-frags
                hfrag[(rt * 16 + kt2) * 512 + l2 * 16 + e2] = hv;   // for phase 3
            }
        }
        wait_async0();        // next step's preS tiles landed
        __syncthreads();      // h_{t+1} fragments + preS visible to all waves
    }
}

// -------------------- phase 3: y = hs @ Wy^T --------------------
__global__ __launch_bounds__(256) void gemm_hy_kernel(const __bf16* __restrict__ hfrag,
                                                      const __bf16* __restrict__ Wyp,
                                                      float* __restrict__ y) {
    int rt = blockIdx.x;
    int s = rt >> 10, t = rt & 1023;
    __shared__ __attribute__((aligned(32))) __bf16 aS[16 * 512];   // 16 KB, K=512
    {
        uint32_t l = lds_addr32(aS) + threadIdx.x * 16;
        uint64_t g = (uint64_t)(uintptr_t)(hfrag + (size_t)rt * 8192) + threadIdx.x * 16;
#pragma unroll
        for (int i = 0; i < 4; ++i)
            async_g2l_b128(l + (uint32_t)i * 4096u, g + (uint64_t)i * 4096u);
        wait_async0();
    }
    __syncthreads();

    int wave = threadIdx.x >> 5, lane = threadIdx.x & 31;
    v16bf a[16];
#pragma unroll
    for (int kt = 0; kt < 16; ++kt) a[kt] = *(const v16bf*)(aS + kt * 512 + lane * 16);

#pragma unroll
    for (int j = 0; j < 2; ++j) {
        int nt = wave * 2 + j;
        v8f acc = {0.f, 0.f, 0.f, 0.f, 0.f, 0.f, 0.f, 0.f};
        const __bf16* bp = Wyp + (size_t)nt * 16 * 512 + (size_t)lane * 16;
#pragma unroll
        for (int kt = 0; kt < 16; ++kt) {
            v16bf b = *(const v16bf*)(bp + (size_t)kt * 512);
            acc = WMMA_BF16(a[kt], b, acc);
        }
#pragma unroll
        for (int r = 0; r < 8; ++r) {
            int m = r + ((lane & 16) ? 8 : 0);
            int bb = s * 16 + m;
            int n = nt * 16 + (lane & 15);
            y[((size_t)bb * T_ + t) * NY_ + n] = acc[r];
        }
    }
}

// -------------------- host launcher --------------------
extern "C" void kernel_launch(void* const* d_in, const int* in_sizes, int n_in,
                              void* d_out, int out_size, void* d_ws, size_t ws_size,
                              hipStream_t stream) {
    const float* x  = (const float*)d_in[0];
    const float* h0 = (const float*)d_in[1];
    const float* Wi = (const float*)d_in[2];
    const float* Wh = (const float*)d_in[3];
    const float* Wy = (const float*)d_in[4];

    float* y  = (float*)d_out;
    float* hs = y + (size_t)B_ * T_ * NY_;

    char* p = (char*)d_ws;
    __bf16* xfrag = (__bf16*)p; p += (size_t)B_ * T_ * NX_ * 2;      //  64 MB
    __bf16* Wip   = (__bf16*)p; p += (size_t)NH_ * NX_ * 2;          // 256 KB
    __bf16* Whp   = (__bf16*)p; p += (size_t)NH_ * NH_ * 2;          // 512 KB
    __bf16* Wyp   = (__bf16*)p; p += (size_t)NY_ * NH_ * 2;          // 256 KB
    float*  Abuf  = (float*)p;  p += (size_t)B_ * T_ * NH_ * 4;      // 256 MB
    __bf16* hfrag = (__bf16*)p;                                      // 128 MB

    pack_x_kernel<<<131072, 256, 0, stream>>>(x, xfrag);
    pack_w_kernel<<<512,  256, 0, stream>>>(Wi, Wip, 8);
    pack_w_kernel<<<1024, 256, 0, stream>>>(Wh, Whp, 16);
    pack_w_kernel<<<512,  256, 0, stream>>>(Wy, Wyp, 16);

    gemm_xwi_kernel<<<8192, 256, 0, stream>>>(xfrag, Wip, Abuf);
    rnn_recurrent_kernel<<<8, 256, 0, stream>>>(Whp, Abuf, h0, hs, hfrag);
    gemm_hy_kernel<<<8192, 256, 0, stream>>>(hfrag, Wyp, y);
}
